// CombineGraph_31516470018772
// MI455X (gfx1250) — matrix operand order repaired and checked by reading, not compiled, not run on hardware
//
#include <hip/hip_runtime.h>
#include <hip/hip_bf16.h>

typedef __attribute__((ext_vector_type(16))) _Float16 v16h;
typedef __attribute__((ext_vector_type(8)))  float    v8f;
typedef __attribute__((ext_vector_type(4)))  unsigned int tdm_v4u;
typedef __attribute__((ext_vector_type(8)))  int          tdm_v8i;
typedef __attribute__((ext_vector_type(4)))  int          tdm_v4i;

#define BATCH 256
#define SDIM  70
#define DDIM  128
#define NSAMP 12
#define SPAD  80    // S padded to 5x16 tiles
#define KPAD  96    // K pad for alpha@H (80 -> 96 = 3x32)
#define K1PAD 160   // K pad for feat@gw1 (129 -> 160 = 5x32)

// ---------------- WMMA tile loaders (contiguous k order, same for A and B) ----------

// A (16x32 f16), row-major source [M][K]: lane holds row m = m0+(l&15),
// K pattern: kbase = k0 + 16*(l>=16); elem e -> K = kbase + e
__device__ __forceinline__ v16h wmma_load_a(const _Float16* A, int lda,
                                            int m0, int k0, int lane) {
  int m  = m0 + (lane & 15);
  int kb = k0 + ((lane >> 4) << 4);
  v16h r;
#pragma unroll
  for (int e = 0; e < 16; ++e) {
    int K = kb + e;
    r[e] = A[m * lda + K];
  }
  return r;
}

// B (32x16 f16) from row-major [K][N]: lane holds col n = n0+(l&15),
// elem e -> K = k0 + 16*(l>=16) + e   (strided reads)
__device__ __forceinline__ v16h wmma_load_b(const _Float16* B, int ldb,
                                            int k0, int n0, int lane) {
  int n  = n0 + (lane & 15);
  int kb = k0 + ((lane >> 4) << 4);
  v16h r;
#pragma unroll
  for (int e = 0; e < 16; ++e) r[e] = B[(kb + e) * ldb + n];
  return r;
}

// B operand from a row-major [N][K] matrix (contiguous 16-half reads)
__device__ __forceinline__ v16h wmma_load_bt(const _Float16* M, int ldm,
                                             int k0, int n0, int lane) {
  int n  = n0 + (lane & 15);
  int kb = k0 + ((lane >> 4) << 4);
  v16h r;
#pragma unroll
  for (int e = 0; e < 16; ++e) r[e] = M[n * ldm + kb + e];
  return r;
}

__device__ __forceinline__ v8f wmma16x16x32(v16h a, v16h b, v8f c) {
  return __builtin_amdgcn_wmma_f32_16x16x32_f16(false, a, false, b,
                                                (short)0, c, false, false);
}

// ---------------- TDM: 2D f16 tile Global -> LDS (ISA 08_async_tensor §8) ---
__device__ __forceinline__ unsigned lds_addr_of(const void* p) {
  return (unsigned)(unsigned long long)
      (__attribute__((address_space(3))) const char*)p;
}

// Copy dim1 rows of dim0 f16 elements (row stride stride0 elements) into LDS.
__device__ __forceinline__ void tdm_load_2d_f16(unsigned lds_addr,
                                                const void* gptr,
                                                unsigned dim0, unsigned dim1,
                                                unsigned stride0) {
  unsigned long long ga = (unsigned long long)gptr;
  tdm_v4u g0;
  g0[0] = 1u;                                        // count=1 (valid user D#)
  g0[1] = lds_addr;                                  // lds_addr[31:0]
  g0[2] = (unsigned)(ga & 0xffffffffull);            // global_addr[31:0]
  g0[3] = (unsigned)((ga >> 32) & 0x1ffffffull)      // global_addr[56:32]
          | (2u << 30);                              // type=2 ("image")
  tdm_v8i g1;
  g1[0] = 0x00010000;                                // data_size=2B
  g1[1] = (int)(dim0 << 16);                         // tensor_dim0[15:0]
  g1[2] = (int)((dim0 >> 16) | (dim1 << 16));        // tdim0 hi | tdim1 lo
  g1[3] = (int)((dim1 >> 16) | (dim0 << 16));        // tdim1 hi | tile_dim0
  g1[4] = (int)(dim1 & 0xffffu);                     // tile_dim1 (tile_dim2=0)
  g1[5] = (int)stride0;                              // tensor_dim0_stride lo
  g1[6] = 0;                                         // stride0 hi | stride1 lo
  g1[7] = 0;
  tdm_v4i z4 = {0, 0, 0, 0};
  tdm_v8i z8 = {0, 0, 0, 0, 0, 0, 0, 0};
  // amdgpu-toolchain (clang-23) 6-arg form: (g0, g1, g2, g3, extra, cpol)
  __builtin_amdgcn_tensor_load_to_lds(g0, g1, z4, z4, z8, 0);
}

// ---------------- weight prep: gw1 -> [N][160] f16 (padded), gw3 -> [N][256] -
__global__ void k_convert(const float* gw1, const float* gw3,
                          _Float16* gw1t, _Float16* gw3t) {
  const int N1 = DDIM * K1PAD;        // 20480
  const int N3 = 2 * DDIM * DDIM;     // 32768
  int idx = blockIdx.x * 256 + threadIdx.x;
  if (idx < N1) {
    int n = idx / K1PAD, k = idx % K1PAD;
    gw1t[idx] = (k < DDIM + 1) ? (_Float16)gw1[k * DDIM + n] : (_Float16)0.f;
  } else if (idx < N1 + N3) {
    int j = idx - N1;
    int n = j >> 8, k = j & 255;
    gw3t[j] = (_Float16)gw3[k * DDIM + n];
  }
}

// ---------------- session mean ----------------------------------------------
__global__ void __launch_bounds__(128)
k_sess(const int* item, const int* mask_item, const float* embedding,
       float* sess) {
  int b = blockIdx.x, d = threadIdx.x;
  float acc = 0.f, cnt = 0.f;
  for (int s = 0; s < SDIM; ++s) {
    float m = (float)mask_item[b * SDIM + s];
    acc += embedding[(long)item[b * SDIM + s] * DDIM + d] * m;
    cnt += m;
  }
  sess[b * DDIM + d] = acc / cnt;
}

// ---------------- local attention + h_local (block per batch) ---------------
__global__ void __launch_bounds__(256)
k_local(const int* inputs, const int* adj, const float* embedding,
        const float* a0, const float* a1, const float* a2, const float* a3,
        float* out_local) {
  __shared__ _Float16 hh[KPAD * DDIM];    // 96x128 f16 (rows >=70 zero)
  __shared__ float    sc[SDIM * SPAD];    // 70x80 f32 scores
  __shared__ _Float16 af[SPAD * KPAD];    // 80x96 f16 softmax probs (padded)
  __shared__ _Float16 ah[4 * DDIM];       // attention vectors f16

  const int b = blockIdx.x;
  const int tid = threadIdx.x;
  const int lane = tid & 31;
  const int wave = tid >> 5;

  for (int idx = tid; idx < KPAD * DDIM; idx += 256) {
    int r = idx >> 7, d = idx & 127;
    float v = 0.f;
    if (r < SDIM) v = embedding[(long)inputs[b * SDIM + r] * DDIM + d];
    hh[idx] = (_Float16)v;
  }
  for (int idx = tid; idx < SPAD * KPAD; idx += 256) af[idx] = (_Float16)0.f;
  for (int idx = tid; idx < 4 * DDIM; idx += 256) {
    int k = idx >> 7, d = idx & 127;
    const float* av = (k == 0) ? a0 : (k == 1) ? a1 : (k == 2) ? a2 : a3;
    ah[idx] = (_Float16)av[d];
  }
  __syncthreads();

  // scores: E_k = (H (.) a_k) @ H^T for k=0..3, fused adj-select + leaky
  for (int t = wave; t < 25; t += 8) {
    int mt = t / 5, nt = t % 5;
    v8f acc0 = {}, acc1 = {}, acc2 = {}, acc3 = {};
    for (int ks = 0; ks < 4; ++ks) {
      int k0 = ks * 32;
      v16h bt = wmma_load_bt(hh, DDIM, k0, nt * 16, lane);
      int m  = mt * 16 + (lane & 15);
      int kb = k0 + ((lane >> 4) << 4);
      v16h A0, A1, A2, A3;
#pragma unroll
      for (int e = 0; e < 16; ++e) {
        int K = kb + e;
        _Float16 hv = hh[m * DDIM + K];
        A0[e] = hv * ah[0 * DDIM + K];
        A1[e] = hv * ah[1 * DDIM + K];
        A2[e] = hv * ah[2 * DDIM + K];
        A3[e] = hv * ah[3 * DDIM + K];
      }
      acc0 = wmma16x16x32(A0, bt, acc0);
      acc1 = wmma16x16x32(A1, bt, acc1);
      acc2 = wmma16x16x32(A2, bt, acc2);
      acc3 = wmma16x16x32(A3, bt, acc3);
    }
#pragma unroll
    for (int v = 0; v < 8; ++v) {
      int i = mt * 16 + ((lane >> 4) << 3) + v;
      int j = nt * 16 + (lane & 15);
      if (i < SDIM) {
        int jj = (j < SDIM) ? j : 0;                 // clamp (avoid OOB read)
        int a  = adj[((long)b * SDIM + i) * SDIM + jj];
        float e = (a == 1) ? acc0[v] : (a == 2) ? acc1[v]
                : (a == 3) ? acc2[v] : acc3[v];
        float lk  = (e > 0.f) ? e : 0.2f * e;        // leaky_relu(0.2)
        float val = (a >= 1 && a <= 4) ? lk : -9.0e15f;
        val = (j < SDIM) ? val : -9.0e15f;
        sc[i * SPAD + j] = val;
      }
    }
  }
  __syncthreads();

  // row softmax -> f16 alpha (padded with zeros)
  if (tid < SDIM) {
    int i = tid;
    float mx = -3.0e38f;
    for (int j = 0; j < SDIM; ++j) mx = fmaxf(mx, sc[i * SPAD + j]);
    float sum = 0.f;
    for (int j = 0; j < SDIM; ++j) sum += expf(sc[i * SPAD + j] - mx);
    float inv = 1.f / sum;
    for (int j = 0; j < KPAD; ++j) {
      float p = (j < SDIM) ? expf(sc[i * SPAD + j] - mx) * inv : 0.f;
      af[i * KPAD + j] = (_Float16)p;
    }
  }
  __syncthreads();

  // h_local = alpha(80x96) @ H(96x128)
  for (int t = wave; t < 40; t += 8) {
    int mt = t / 8, nt = t % 8;
    v8f acc = {};
    for (int ks = 0; ks < 3; ++ks) {
      v16h A  = wmma_load_a(af, KPAD, mt * 16, ks * 32, lane);
      v16h Bt = wmma_load_b(hh, DDIM, ks * 32, nt * 16, lane);
      acc = wmma16x16x32(A, Bt, acc);
    }
#pragma unroll
    for (int v = 0; v < 8; ++v) {
      int i = mt * 16 + ((lane >> 4) << 3) + v;
      int d = nt * 16 + (lane & 15);
      if (i < SDIM) out_local[((long)b * SDIM + i) * DDIM + d] = acc[v];
    }
  }
}

// ---------------- neighbor gating (block per (b,s)) -------------------------
__global__ void __launch_bounds__(128)
k_global(const int* inputs, const int* adj_all, const float* num_w,
         const float* embedding, const float* gw2, const float* sess,
         const _Float16* gw1t, _Float16* naggh) {
  __shared__ _Float16 w1t[DDIM * K1PAD];  // 128x160 gw1^T (padded), TDM dest
  __shared__ _Float16 fA[16 * K1PAD];     // 16x160 feat tile
  __shared__ float    al[16 * DDIM];      // leaky(feat@gw1)
  __shared__ float    nv[NSAMP * DDIM];   // neighbor embeddings
  __shared__ float    se[DDIM];
  __shared__ float    scn[NSAMP];

  const int blk  = blockIdx.x;
  const int b    = blk / SDIM, s = blk % SDIM;
  const int tid  = threadIdx.x;
  const int lane = tid & 31, wave = tid >> 5;
  const int node = inputs[b * SDIM + s];

  // Wave 0 kicks off the TDM DMA of the 128x160 f16 weight panel into LDS;
  // the gather loads below overlap with it.
  if (wave == 0) {
    tdm_load_2d_f16(lds_addr_of(w1t), gw1t, K1PAD, DDIM, K1PAD);
  }

  se[tid] = sess[b * DDIM + tid];
  for (int idx = tid; idx < NSAMP * DDIM; idx += 128) {
    int n = idx >> 7, d = idx & 127;
    nv[idx] = embedding[(long)adj_all[node * NSAMP + n] * DDIM + d];
  }
  __syncthreads();
  for (int idx = tid; idx < 16 * K1PAD; idx += 128) {
    int n = idx / K1PAD, c = idx % K1PAD;
    float v = 0.f;
    if (n < NSAMP) {
      if (c < DDIM)       v = se[c] * nv[n * DDIM + c];
      else if (c == DDIM) v = num_w[node * NSAMP + n];
    }
    fA[idx] = (_Float16)v;
  }
  __builtin_amdgcn_s_wait_tensorcnt((unsigned short)0);
  __syncthreads();

  for (int q = 0; q < 2; ++q) {
    int nt = wave * 2 + q;
    v8f acc = {};
    for (int ks = 0; ks < 5; ++ks) {
      v16h A = wmma_load_a(fA, K1PAD, 0, ks * 32, lane);
      v16h B = wmma_load_bt(w1t, K1PAD, ks * 32, nt * 16, lane);
      acc = wmma16x16x32(A, B, acc);
    }
#pragma unroll
    for (int v = 0; v < 8; ++v) {
      int r = ((lane >> 4) << 3) + v;
      int c = nt * 16 + (lane & 15);
      float x = acc[v];
      al[r * DDIM + c] = (x > 0.f) ? x : 0.2f * x;
    }
  }
  __syncthreads();
  if (tid < NSAMP) {
    float sum = 0.f;
    for (int d = 0; d < DDIM; ++d) sum += al[tid * DDIM + d] * gw2[d];
    scn[tid] = sum;
  }
  __syncthreads();
  {
    int d = tid;
    float mx = -3.0e38f;
    for (int n = 0; n < NSAMP; ++n) mx = fmaxf(mx, scn[n]);
    float ssum = 0.f;
    for (int n = 0; n < NSAMP; ++n) ssum += expf(scn[n] - mx);
    float inv = 1.f / ssum;
    float agg = 0.f;
    for (int n = 0; n < NSAMP; ++n)
      agg += expf(scn[n] - mx) * inv * nv[n * DDIM + d];
    naggh[(long)blk * DDIM + d] = (_Float16)agg;
  }
}

// ---------------- final projection (block per 64 rows) ----------------------
__global__ void __launch_bounds__(256)
k_final(const int* inputs, const float* embedding, const _Float16* naggh,
        const _Float16* gw3t, const float* gbias, float* out_global) {
  __shared__ _Float16 Ah[64 * 256];   // [h | nagg]
  __shared__ float    bi[DDIM];
  const int blk  = blockIdx.x;
  const int tid  = threadIdx.x, lane = tid & 31, wave = tid >> 5;
  if (tid < DDIM) bi[tid] = gbias[tid];
  for (int idx = tid; idx < 64 * 256; idx += 256) {
    int r = idx >> 8, c = idx & 255;
    long grow = (long)blk * 64 + r;
    float v;
    if (c < DDIM) v = embedding[(long)inputs[grow] * DDIM + c];
    else          v = (float)naggh[grow * DDIM + (c - DDIM)];
    Ah[idx] = (_Float16)v;
  }
  __syncthreads();
  const int nt = wave;
  v8f acc[4] = { {}, {}, {}, {} };
  for (int ks = 0; ks < 8; ++ks) {
    v16h B = wmma_load_bt(gw3t, 256, ks * 32, nt * 16, lane);  // contiguous
#pragma unroll
    for (int mt = 0; mt < 4; ++mt) {
      v16h A = wmma_load_a(Ah, 256, mt * 16, ks * 32, lane);
      acc[mt] = wmma16x16x32(A, B, acc[mt]);
    }
  }
#pragma unroll
  for (int mt = 0; mt < 4; ++mt)
#pragma unroll
    for (int v = 0; v < 8; ++v) {
      int r = mt * 16 + ((lane >> 4) << 3) + v;
      int c = nt * 16 + (lane & 15);
      long grow = (long)blk * 64 + r;
      float x = acc[mt][v] + bi[c];
      out_global[grow * DDIM + c] = (x > 0.f) ? x : 0.f;
    }
}

// ---------------- launcher ---------------------------------------------------
extern "C" void kernel_launch(void* const* d_in, const int* in_sizes, int n_in,
                              void* d_out, int out_size, void* d_ws, size_t ws_size,
                              hipStream_t stream) {
  (void)in_sizes; (void)n_in; (void)out_size; (void)ws_size;
  const int*   inputs    = (const int*)d_in[0];
  const int*   adj       = (const int*)d_in[1];
  const int*   mask_item = (const int*)d_in[2];
  const int*   item      = (const int*)d_in[3];
  /* d_in[4] input_times: unused by reference output */
  const int*   adj_all   = (const int*)d_in[5];
  const float* num_w     = (const float*)d_in[6];
  const float* embedding = (const float*)d_in[7];
  const float* a0        = (const float*)d_in[8];
  const float* a1        = (const float*)d_in[9];
  const float* a2        = (const float*)d_in[10];
  const float* a3        = (const float*)d_in[11];
  const float* gw1       = (const float*)d_in[12];
  const float* gw2       = (const float*)d_in[13];
  const float* gw3       = (const float*)d_in[14];
  const float* gbias     = (const float*)d_in[15];

  float* out_local  = (float*)d_out;
  float* out_global = out_local + (long)BATCH * SDIM * DDIM;

  char* wsb = (char*)d_ws;
  float*    ws_sess  = (float*)wsb;                               // 131072 B
  _Float16* ws_gw1t  = (_Float16*)(wsb + 131072);                 //  40960 B
  _Float16* ws_gw3t  = (_Float16*)(wsb + 131072 + 40960);         //  65536 B
  _Float16* ws_naggh = (_Float16*)(wsb + 131072 + 40960 + 65536); // 4.59 MB

  k_convert<<<(DDIM * K1PAD + 2 * DDIM * DDIM + 255) / 256, 256, 0, stream>>>(
      gw1, gw3, ws_gw1t, ws_gw3t);
  k_sess<<<BATCH, 128, 0, stream>>>(item, mask_item, embedding, ws_sess);
  k_local<<<BATCH, 256, 0, stream>>>(inputs, adj, embedding,
                                     a0, a1, a2, a3, out_local);
  k_global<<<BATCH * SDIM, 128, 0, stream>>>(inputs, adj_all, num_w, embedding,
                                             gw2, ws_sess, ws_gw1t, ws_naggh);
  k_final<<<(BATCH * SDIM) / 64, 256, 0, stream>>>(inputs, embedding, ws_naggh,
                                                   ws_gw3t, gbias, out_global);
}
